// MoE_4380866642221
// MI455X (gfx1250) — matrix operand order repaired and checked by reading, not compiled, not run on hardware
//
#include <hip/hip_runtime.h>
#include <cstdint>
#include <cstddef>

// ---------------------------------------------------------------------------
// Types for CDNA5 WMMA (gfx1250, wave32)
// ---------------------------------------------------------------------------
typedef __attribute__((ext_vector_type(16))) __bf16 v16bf;
typedef __attribute__((ext_vector_type(8)))  float  v8f;

#define LSTR 40   // LDS row stride in ushorts: 32 data + 8 pad = 80B (conflict-free b128 lanes)

// Async global->LDS (CDNA5 GLOBAL_LOAD_ASYNC_TO_LDS_B128, ASYNCcnt-tracked)
#if defined(__has_builtin)
#if __has_builtin(__builtin_amdgcn_global_load_async_to_lds_b128) && \
    __has_builtin(__builtin_amdgcn_s_wait_asynccnt)
#define HAVE_ASYNC_LDS 1
#endif
#endif
#ifndef HAVE_ASYNC_LDS
#define HAVE_ASYNC_LDS 0
#endif

#if HAVE_ASYNC_LDS
typedef int v4i_vs __attribute__((vector_size(16)));
typedef __attribute__((address_space(1))) v4i_vs* g_v4i_ptr;
typedef __attribute__((address_space(3))) v4i_vs* l_v4i_ptr;
__device__ __forceinline__ void async_cp16(const unsigned short* g, unsigned short* l) {
  __builtin_amdgcn_global_load_async_to_lds_b128((g_v4i_ptr)(g), (l_v4i_ptr)(l), 0, 0);
}
#endif

union FragU { uint4 u[2]; v16bf v; };

__device__ __forceinline__ unsigned short f2bf(float f) {
  unsigned int u = __float_as_uint(f);
  u += 0x7FFFu + ((u >> 16) & 1u);          // round-to-nearest-even
  return (unsigned short)(u >> 16);
}

__device__ __forceinline__ v8f wmma_bf16(v16bf a, v16bf b, v8f c) {
  return __builtin_amdgcn_wmma_f32_16x16x32_bf16(
      /*neg_a=*/false, a, /*neg_b=*/false, b,
      /*c_mod=*/(short)0, c, /*reuse_a=*/false, /*reuse_b=*/false);
}

// A fragment (16x32 bf16, MxK): lane<16 -> row=lane, K={0..7,16..23};
// lane>=16 -> row=lane-16, K={8..15,24..31}. LDS tile is [M][K] row-major.
__device__ __forceinline__ v16bf frag_a(const unsigned short* lds, int rowBase, int lane) {
  int l = lane & 15;
  int hi = (lane >> 4) << 3;                // 0 or 8 ushorts
  const unsigned short* p = lds + (rowBase + l) * LSTR + hi;
  FragU f;
  f.u[0] = *(const uint4*)(p);              // K group 0 (8 bf16)
  f.u[1] = *(const uint4*)(p + 16);         // K group +16
  return f.v;
}

// B fragment (32x16 bf16, KxN): lane<16 -> col=lane, K={0..7,16..23}; lane>=16 -> K={8..15,24..31}.
// LDS tile is [N][K] (i.e., weight row-major W[n][k]).
__device__ __forceinline__ v16bf frag_b(const unsigned short* lds, int colBase, int lane) {
  int l = lane & 15;
  int hi = (lane >> 4) << 3;                // 0 or 8 ushorts
  const unsigned short* p = lds + (colBase + l) * LSTR + hi;
  FragU f;
  f.u[0] = *(const uint4*)(p);
  f.u[1] = *(const uint4*)(p + 16);
  return f.v;
}

// ---------------------------------------------------------------------------
// fp32 -> bf16 conversion kernels
// ---------------------------------------------------------------------------
__global__ __launch_bounds__(256) void cvt_bf16_kernel(
    const float* __restrict__ s, unsigned short* __restrict__ d, int n) {
  int i = (blockIdx.x * 256 + threadIdx.x) * 4;
  if (i + 3 >= n) return;
  float4 v = *(const float4*)(s + i);
  unsigned int lo = (unsigned int)f2bf(v.x) | ((unsigned int)f2bf(v.y) << 16);
  unsigned int hi = (unsigned int)f2bf(v.z) | ((unsigned int)f2bf(v.w) << 16);
  uint2 o; o.x = lo; o.y = hi;
  *(uint2*)(d + i) = o;
}

// rows x Hsrc  ->  rows x Hp (zero-padded tail)
__global__ __launch_bounds__(256) void cvt_pad_kernel(
    const float* __restrict__ s, unsigned short* __restrict__ d,
    int rows, int Hsrc, int Hp) {
  int i = blockIdx.x * 256 + threadIdx.x;
  if (i >= rows * Hp) return;
  int r = i / Hp, h = i - r * Hp;
  d[i] = (h < Hsrc) ? f2bf(s[(size_t)r * Hsrc + h]) : (unsigned short)0;
}

// ---------------------------------------------------------------------------
// Router: one wave per token. counters: [0..7]=pair counts, [8..15]=top1 counts,
// [16..24]=offsets, [25..32]=cursor.
// ---------------------------------------------------------------------------
__global__ __launch_bounds__(256) void router_kernel(
    const float* __restrict__ x, const float* __restrict__ gate_w,
    float* __restrict__ probs, int* __restrict__ top_idx, float* __restrict__ top_w,
    int* __restrict__ counters, int S, int C) {
  int tok  = (blockIdx.x * 256 + threadIdx.x) >> 5;
  int lane = threadIdx.x & 31;
  if (tok >= S) return;
  const float* xr = x + (size_t)tok * C;

  float p[8];
#pragma unroll
  for (int e = 0; e < 8; ++e) {
    const float* g = gate_w + e * C;
    float s = 0.f;
#pragma unroll
    for (int i = 0; i < 4; ++i) {
      int c = lane * 4 + i * 128;
      float4 xv = *(const float4*)(xr + c);
      float4 gv = *(const float4*)(g + c);
      s += xv.x * gv.x + xv.y * gv.y + xv.z * gv.z + xv.w * gv.w;
    }
#pragma unroll
    for (int off = 16; off >= 1; off >>= 1) s += __shfl_xor(s, off, 32);
    p[e] = s;
  }
  // softmax (fp32)
  float m = p[0];
#pragma unroll
  for (int e = 1; e < 8; ++e) m = fmaxf(m, p[e]);
  float sum = 0.f;
#pragma unroll
  for (int e = 0; e < 8; ++e) { p[e] = __expf(p[e] - m); sum += p[e]; }
  float inv = 1.f / sum;
#pragma unroll
  for (int e = 0; e < 8; ++e) p[e] *= inv;

  // top-2 (ties -> lower index, matching lax.top_k)
  int i0 = 0; float b0 = p[0];
#pragma unroll
  for (int e = 1; e < 8; ++e) if (p[e] > b0) { b0 = p[e]; i0 = e; }
  int i1 = (i0 == 0) ? 1 : 0; float b1 = p[i1];
#pragma unroll
  for (int e = 0; e < 8; ++e) if (e != i0 && p[e] > b1) { b1 = p[e]; i1 = e; }

  if (lane == 0) {
#pragma unroll
    for (int e = 0; e < 8; ++e) probs[(size_t)tok * 8 + e] = p[e];
    float sw = b0 + b1;
    top_idx[tok * 2 + 0] = i0;  top_idx[tok * 2 + 1] = i1;
    top_w[tok * 2 + 0]   = b0 / sw;
    top_w[tok * 2 + 1]   = b1 / sw;
    atomicAdd(&counters[i0], 1);
    atomicAdd(&counters[i1], 1);
    atomicAdd(&counters[8 + i0], 1);
  }
}

__global__ void zero_counters_kernel(int* c, unsigned short* z) {
  int t = threadIdx.x;
  if (t < 40) c[t] = 0;
  if (t < 64) z[t] = 0;
}

__global__ void scan_kernel(int* c) {
  if (threadIdx.x == 0) {
    int acc = 0;
    for (int e = 0; e < 8; ++e) { c[16 + e] = acc; acc += c[e]; c[25 + e] = 0; }
    c[24] = acc;
  }
}

__global__ __launch_bounds__(256) void scatter_kernel(
    const int* __restrict__ top_idx, int* __restrict__ counters,
    int* __restrict__ pair_token, int* __restrict__ pair_slot, int S) {
  int s = blockIdx.x * 256 + threadIdx.x;
  if (s >= S) return;
#pragma unroll
  for (int k = 0; k < 2; ++k) {
    int e = top_idx[s * 2 + k];
    int pos = atomicAdd(&counters[25 + e], 1);
    int p = counters[16 + e] + pos;
    pair_token[p] = s;
    pair_slot[s * 2 + k] = p;
  }
}

// deterministic per-expert importance sums
__global__ __launch_bounds__(256) void imp_reduce_kernel(
    const float* __restrict__ probs, float* __restrict__ imp, int S) {
  int e = blockIdx.x;
  __shared__ float sm[256];
  float s = 0.f;
  for (int i = threadIdx.x; i < S; i += 256) s += probs[(size_t)i * 8 + e];
  sm[threadIdx.x] = s;
  __syncthreads();
  for (int st = 128; st > 0; st >>= 1) {
    if (threadIdx.x < st) sm[threadIdx.x] += sm[threadIdx.x + st];
    __syncthreads();
  }
  if (threadIdx.x == 0) imp[e] = sm[0];
}

__global__ void finalize_aux_kernel(const float* __restrict__ imp,
                                    const int* __restrict__ counters,
                                    float* __restrict__ out_aux, int S, int E) {
  if (threadIdx.x == 0) {
    float a = 0.f;
    float invS = 1.f / (float)S;
    for (int e = 0; e < E; ++e)
      a += (imp[e] * invS) * ((float)counters[8 + e] * invS);
    out_aux[0] = (float)E * a;
  }
}

// ---------------------------------------------------------------------------
// GEMM1: P_h[p, :] = silu(x_tok W1^T) * (x_tok W3^T), bf16 out.
// WG tile 128(M) x 64(N); 8 waves as 4x2; wave tile 32x32 for BOTH h1 and h3.
// Double-buffered LDS; async global->LDS fills overlap WMMA (1 barrier/k-step).
// ---------------------------------------------------------------------------
__global__ __launch_bounds__(256) void gemm1_kernel(
    const unsigned short* __restrict__ xb,        // [S][C] bf16
    const int* __restrict__ pair_token,           // null => identity rows
    const int* __restrict__ counters,             // null => fixedCount; else offsets[16+e]
    int fixedCount,
    const unsigned short* __restrict__ w1b,       // [E][H][C] bf16
    const unsigned short* __restrict__ w3b,
    const unsigned short* __restrict__ zsrc,      // >=64B of zeros (OOB lane source)
    unsigned short* __restrict__ ph,              // [rowspace][Hp] bf16
    int C, int H, int Hp) {
  int e = blockIdx.z;
  int base = 0, count = fixedCount;
  if (counters) { base = counters[16 + e]; count = counters[17 + e] - base; }
  int m0 = blockIdx.x * 128;
  if (m0 >= count) return;
  int n0 = blockIdx.y * 64;

  __shared__ unsigned short lA [2][128 * LSTR];
  __shared__ unsigned short lB1[2][ 64 * LSTR];
  __shared__ unsigned short lB3[2][ 64 * LSTR];

  int tid = threadIdx.x;
  int lane = tid & 31;
  int w = tid >> 5;
  int wm = (w >> 1) * 32;
  int wn = (w & 1) * 32;

  // A tile loader: 2 threads/row, 16 bf16 each
  int ar = tid >> 1, ah = (tid & 1) * 16;
  int arl = m0 + ar;
  int tok;
  if (pair_token) tok = pair_token[base + ((arl < count) ? arl : 0)];
  else            tok = arl;
  const unsigned short* aRow = xb + (size_t)tok * C;

  // B tile loader: 4 threads/row, 8 bf16 each (per matrix)
  int br = tid >> 2, bq = (tid & 3) * 8;
  int nglob = n0 + br;
  bool bOK = nglob < H;
  size_t wOff = ((size_t)e * H + (size_t)(bOK ? nglob : 0)) * C;

  v8f z8 = {0.f, 0.f, 0.f, 0.f, 0.f, 0.f, 0.f, 0.f};
  v8f acc1[2][2], acc3[2][2];
#pragma unroll
  for (int i = 0; i < 2; ++i)
#pragma unroll
    for (int j = 0; j < 2; ++j) { acc1[i][j] = z8; acc3[i][j] = z8; }

  auto issue = [&](int ks, int buf) {
    int k0 = ks << 5;
    const unsigned short* a0  = aRow + k0 + ah;
    const unsigned short* b1p = bOK ? (w1b + wOff + k0 + bq) : zsrc;
    const unsigned short* b3p = bOK ? (w3b + wOff + k0 + bq) : zsrc;
#if HAVE_ASYNC_LDS
    async_cp16(a0,     &lA [buf][ar * LSTR + ah]);
    async_cp16(a0 + 8, &lA [buf][ar * LSTR + ah + 8]);
    async_cp16(b1p,    &lB1[buf][br * LSTR + bq]);
    async_cp16(b3p,    &lB3[buf][br * LSTR + bq]);
#else
    *(uint4*)&lA [buf][ar * LSTR + ah]     = *(const uint4*)a0;
    *(uint4*)&lA [buf][ar * LSTR + ah + 8] = *(const uint4*)(a0 + 8);
    *(uint4*)&lB1[buf][br * LSTR + bq]     = *(const uint4*)b1p;
    *(uint4*)&lB3[buf][br * LSTR + bq]     = *(const uint4*)b3p;
#endif
  };

  int kSteps = C >> 5;
  issue(0, 0);
  for (int ks = 0; ks < kSteps; ++ks) {
    int buf = ks & 1;
#if HAVE_ASYNC_LDS
    __builtin_amdgcn_s_wait_asynccnt(0);    // stage ks LDS fill complete (this wave)
#endif
    __syncthreads();                        // visible to all; prior reads of buf^1 done
    if (ks + 1 < kSteps) issue(ks + 1, buf ^ 1);

    v16bf a0  = frag_a(lA [buf], wm,      lane);
    v16bf a1  = frag_a(lA [buf], wm + 16, lane);
    v16bf b10 = frag_b(lB1[buf], wn,      lane);
    v16bf b11 = frag_b(lB1[buf], wn + 16, lane);
    v16bf b30 = frag_b(lB3[buf], wn,      lane);
    v16bf b31 = frag_b(lB3[buf], wn + 16, lane);

    acc1[0][0] = wmma_bf16(a0, b10, acc1[0][0]);
    acc1[0][1] = wmma_bf16(a0, b11, acc1[0][1]);
    acc1[1][0] = wmma_bf16(a1, b10, acc1[1][0]);
    acc1[1][1] = wmma_bf16(a1, b11, acc1[1][1]);
    acc3[0][0] = wmma_bf16(a0, b30, acc3[0][0]);
    acc3[0][1] = wmma_bf16(a0, b31, acc3[0][1]);
    acc3[1][0] = wmma_bf16(a1, b30, acc3[1][0]);
    acc3[1][1] = wmma_bf16(a1, b31, acc3[1][1]);
  }

  int l15 = lane & 15;
  int lhi = (lane >> 4) * 8;
#pragma unroll
  for (int i = 0; i < 2; ++i) {
#pragma unroll
    for (int j = 0; j < 2; ++j) {
      int colg = n0 + wn + j * 16 + l15;
      if (colg < Hp) {
#pragma unroll
        for (int r = 0; r < 8; ++r) {
          int rowl = m0 + wm + i * 16 + lhi + r;
          if (rowl < count) {
            float h1 = acc1[i][j][r];
            float h3 = acc3[i][j][r];
            float sv = h1 / (1.f + __expf(-h1));
            ph[(size_t)(base + rowl) * Hp + colg] = f2bf(sv * h3);
          }
        }
      }
    }
  }
}

// ---------------------------------------------------------------------------
// GEMM2: Y[p, :] = P_h[p, :] @ W2^T   (K=Hp zero-padded). WG 128x128, waves 2x4,
// wave tile 64x32 (8 accumulators). Double-buffered, async fills.
// ---------------------------------------------------------------------------
__global__ __launch_bounds__(256) void gemm2_kernel(
    const unsigned short* __restrict__ A,         // [rowspace][Hp] bf16
    const int* __restrict__ counters, int fixedCount,
    const unsigned short* __restrict__ w2b,       // [E][C][Hp] bf16
    float* __restrict__ Y,                        // [rowspace][C] f32
    int C, int Hp) {
  int e = blockIdx.z;
  int base = 0, count = fixedCount;
  if (counters) { base = counters[16 + e]; count = counters[17 + e] - base; }
  int m0 = blockIdx.x * 128;
  if (m0 >= count) return;
  int n0 = blockIdx.y * 128;

  __shared__ unsigned short lA[2][128 * LSTR];
  __shared__ unsigned short lB[2][128 * LSTR];

  int tid = threadIdx.x, lane = tid & 31, w = tid >> 5;
  int wm = (w >> 2) * 64, wn = (w & 3) * 32;

  int ar = tid >> 1, ah = (tid & 1) * 16;
  const unsigned short* aRow = A + (size_t)(base + m0 + ar) * Hp;     // rowspace padded
  const unsigned short* bRow = w2b + ((size_t)e * C + (size_t)(n0 + ar)) * Hp;

  v8f z8 = {0.f, 0.f, 0.f, 0.f, 0.f, 0.f, 0.f, 0.f};
  v8f acc[4][2];
#pragma unroll
  for (int i = 0; i < 4; ++i)
#pragma unroll
    for (int j = 0; j < 2; ++j) acc[i][j] = z8;

  auto issue = [&](int ks, int buf) {
    int k0 = ks << 5;
    const unsigned short* ap = aRow + k0 + ah;
    const unsigned short* bp = bRow + k0 + ah;
#if HAVE_ASYNC_LDS
    async_cp16(ap,     &lA[buf][ar * LSTR + ah]);
    async_cp16(ap + 8, &lA[buf][ar * LSTR + ah + 8]);
    async_cp16(bp,     &lB[buf][ar * LSTR + ah]);
    async_cp16(bp + 8, &lB[buf][ar * LSTR + ah + 8]);
#else
    *(uint4*)&lA[buf][ar * LSTR + ah]     = *(const uint4*)ap;
    *(uint4*)&lA[buf][ar * LSTR + ah + 8] = *(const uint4*)(ap + 8);
    *(uint4*)&lB[buf][ar * LSTR + ah]     = *(const uint4*)bp;
    *(uint4*)&lB[buf][ar * LSTR + ah + 8] = *(const uint4*)(bp + 8);
#endif
  };

  int kSteps = Hp >> 5;
  issue(0, 0);
  for (int ks = 0; ks < kSteps; ++ks) {
    int buf = ks & 1;
#if HAVE_ASYNC_LDS
    __builtin_amdgcn_s_wait_asynccnt(0);
#endif
    __syncthreads();
    if (ks + 1 < kSteps) issue(ks + 1, buf ^ 1);

    v16bf af[4], bf[2];
#pragma unroll
    for (int i = 0; i < 4; ++i) af[i] = frag_a(lA[buf], wm + i * 16, lane);
#pragma unroll
    for (int j = 0; j < 2; ++j) bf[j] = frag_b(lB[buf], wn + j * 16, lane);
#pragma unroll
    for (int i = 0; i < 4; ++i)
#pragma unroll
      for (int j = 0; j < 2; ++j) acc[i][j] = wmma_bf16(af[i], bf[j], acc[i][j]);
  }

  int l15 = lane & 15;
  int lhi = (lane >> 4) * 8;
#pragma unroll
  for (int i = 0; i < 4; ++i) {
#pragma unroll
    for (int j = 0; j < 2; ++j) {
      int colg = n0 + wn + j * 16 + l15;
#pragma unroll
      for (int r = 0; r < 8; ++r) {
        int rowl = m0 + wm + i * 16 + lhi + r;
        if (rowl < count)
          Y[(size_t)(base + rowl) * C + colg] = acc[i][j][r];
      }
    }
  }
}

// ---------------------------------------------------------------------------
// Deterministic combine: out = w0*y[slot0] + w1*y[slot1] + shared
// ---------------------------------------------------------------------------
__global__ __launch_bounds__(256) void combine_kernel(
    const float* __restrict__ ypairs, const float* __restrict__ shy,
    const int* __restrict__ pair_slot, const float* __restrict__ top_w,
    float* __restrict__ out, int S, int C) {
  int idx = blockIdx.x * 256 + threadIdx.x;
  int c4 = C >> 2;
  if (idx >= S * c4) return;
  int s = idx / c4;
  int c = (idx - s * c4) << 2;
  int p0 = pair_slot[s * 2 + 0], p1 = pair_slot[s * 2 + 1];
  float w0 = top_w[s * 2 + 0],   w1 = top_w[s * 2 + 1];
  float4 a  = *(const float4*)(ypairs + (size_t)p0 * C + c);
  float4 b  = *(const float4*)(ypairs + (size_t)p1 * C + c);
  float4 sh = *(const float4*)(shy + (size_t)s * C + c);
  float4 o;
  o.x = w0 * a.x + w1 * b.x + sh.x;
  o.y = w0 * a.y + w1 * b.y + sh.y;
  o.z = w0 * a.z + w1 * b.z + sh.z;
  o.w = w0 * a.w + w1 * b.w + sh.w;
  *(float4*)(out + (size_t)s * C + c) = o;
}

// ---------------------------------------------------------------------------
// Host launcher
// ---------------------------------------------------------------------------
extern "C" void kernel_launch(void* const* d_in, const int* in_sizes, int n_in,
                              void* d_out, int out_size, void* d_ws, size_t ws_size,
                              hipStream_t stream) {
  (void)in_sizes; (void)n_in; (void)out_size; (void)ws_size;
  const int S = 16384, C = 512, E = 8, H = 1372, Hp = 1376;
  const int P = 2 * S, Ppad = P + 128;

  const float* x    = (const float*)d_in[0];
  const float* gw   = (const float*)d_in[1];
  const float* w1   = (const float*)d_in[2];
  const float* w3   = (const float*)d_in[3];
  const float* w2   = (const float*)d_in[4];
  const float* sw1  = (const float*)d_in[5];
  const float* sw3  = (const float*)d_in[6];
  const float* sw2  = (const float*)d_in[7];
  float* out = (float*)d_out;

  char* ws = (char*)d_ws;
  size_t cur = 0;
  auto alloc = [&](size_t bytes) -> char* {
    char* p = ws + cur;
    cur = (cur + bytes + 255) & ~(size_t)255;
    return p;
  };

  unsigned short* xb   = (unsigned short*)alloc((size_t)S * C * 2);
  unsigned short* w1b  = (unsigned short*)alloc((size_t)E * H * C * 2);
  unsigned short* w3b  = (unsigned short*)alloc((size_t)E * H * C * 2);
  unsigned short* w2b  = (unsigned short*)alloc((size_t)E * C * Hp * 2);
  unsigned short* sw1b = (unsigned short*)alloc((size_t)H * C * 2);
  unsigned short* sw3b = (unsigned short*)alloc((size_t)H * C * 2);
  unsigned short* sw2b = (unsigned short*)alloc((size_t)C * Hp * 2);
  float* probs = (float*)alloc((size_t)S * 8 * 4);
  int*   topi  = (int*)  alloc((size_t)S * 2 * 4);
  float* topw  = (float*)alloc((size_t)S * 2 * 4);
  int*   ptok  = (int*)  alloc((size_t)Ppad * 4);
  int*   pslot = (int*)  alloc((size_t)S * 2 * 4);
  int*   cnt   = (int*)  alloc(64 * 4);
  float* imp   = (float*)alloc(8 * 4);
  unsigned short* zsrc = (unsigned short*)alloc(64 * 2);   // zeroed OOB source
  unsigned short* ph  = (unsigned short*)alloc((size_t)Ppad * Hp * 2);
  unsigned short* shh = (unsigned short*)alloc((size_t)S * Hp * 2);
  float* yp  = (float*)alloc((size_t)Ppad * C * 4);
  float* shy = (float*)alloc((size_t)S * C * 4);

  // 0) zero counters + zero-source region
  zero_counters_kernel<<<1, 64, 0, stream>>>(cnt, zsrc);

  // 1) conversions (all counts are multiples of 1024)
  cvt_bf16_kernel<<<(S * C) / 1024, 256, 0, stream>>>(x, xb, S * C);
  cvt_bf16_kernel<<<(E * H * C) / 1024, 256, 0, stream>>>(w1, w1b, E * H * C);
  cvt_bf16_kernel<<<(E * H * C) / 1024, 256, 0, stream>>>(w3, w3b, E * H * C);
  cvt_bf16_kernel<<<(H * C) / 1024, 256, 0, stream>>>(sw1, sw1b, H * C);
  cvt_bf16_kernel<<<(H * C) / 1024, 256, 0, stream>>>(sw3, sw3b, H * C);
  cvt_pad_kernel<<<(E * C * Hp + 255) / 256, 256, 0, stream>>>(w2, w2b, E * C, H, Hp);
  cvt_pad_kernel<<<(C * Hp + 255) / 256, 256, 0, stream>>>(sw2, sw2b, C, H, Hp);

  // 2) router (8 waves/block, 1 wave per token)
  router_kernel<<<S / 8, 256, 0, stream>>>(x, gw, probs, topi, topw, cnt, S, C);

  // 3) scan + scatter
  scan_kernel<<<1, 32, 0, stream>>>(cnt);
  scatter_kernel<<<S / 256, 256, 0, stream>>>(topi, cnt, ptok, pslot, S);

  // 4) aux loss (deterministic reduce)
  imp_reduce_kernel<<<8, 256, 0, stream>>>(probs, imp, S);
  finalize_aux_kernel<<<1, 32, 0, stream>>>(imp, cnt, out + (size_t)S * C, S, E);

  // 5) expert GEMM1 (sparse, gathered rows) + shared GEMM1
  gemm1_kernel<<<dim3(P / 128 / 2, (Hp + 63) / 64, E), 256, 0, stream>>>(
      xb, ptok, cnt, 0, w1b, w3b, zsrc, ph, C, H, Hp);
  gemm1_kernel<<<dim3(S / 128, (Hp + 63) / 64, 1), 256, 0, stream>>>(
      xb, nullptr, nullptr, S, sw1b, sw3b, zsrc, shh, C, H, Hp);

  // 6) expert GEMM2 + shared GEMM2
  gemm2_kernel<<<dim3(P / 128 / 2, C / 128, E), 256, 0, stream>>>(
      ph, cnt, 0, w2b, yp, C, Hp);
  gemm2_kernel<<<dim3(S / 128, C / 128, 1), 256, 0, stream>>>(
      shh, nullptr, S, sw2b, shy, C, Hp);

  // 7) combine
  combine_kernel<<<(S * (C / 4)) / 256, 256, 0, stream>>>(
      yp, shy, pslot, topw, out, S, C);
}
